// RGCNModel_55284819034409
// MI455X (gfx1250) — matrix operand order repaired and checked by reading, not compiled, not run on hardware
//
#include <hip/hip_runtime.h>
#include <hip/hip_bf16.h>
#include <stdint.h>
#include <stddef.h>

// ---------------------------------------------------------------------------
// RGCN, 4 layers + FC/sigmoid, for MI455X (gfx1250, wave32).
// Strategy: aggregate-then-transform (linearity of segment_sum + matmul),
// f32 atomic scatter into L2-resident buffers, bf16 WMMA GEMMs (f32 acc),
// each wave register-blocked over a 16x64 output strip (4 WMMA tiles).
// ---------------------------------------------------------------------------

typedef __bf16 v16bf __attribute__((ext_vector_type(16)));
typedef float  v8f   __attribute__((ext_vector_type(8)));

#define NODES  50000
#define MTILES (NODES / 16)   // 3125, exact

// ---------------------------------------------------------------------- zero
__global__ void zero4_kernel(float4* __restrict__ p, int n4) {
    int i  = blockIdx.x * blockDim.x + threadIdx.x;
    int st = gridDim.x * blockDim.x;
    float4 z = {0.f, 0.f, 0.f, 0.f};
    for (; i < n4; i += st) p[i] = z;
}

// ------------------------------------------------------------- weight pack
// Pack W[Fin][Fout] (f32 row-major) into per-lane WMMA B-fragment order, bf16.
// Packed index: ((kt*NT + nt)*512) + lane*16 + j
//   K = kt*32 + (lane<16 ? 0 : 8) + j + (j<8 ? 0 : 8),  N = nt*16 + (lane&15)
__global__ void pack_w_kernel(const float* __restrict__ W, __bf16* __restrict__ P,
                              int Fin, int Fout) {
    int idx   = blockIdx.x * blockDim.x + threadIdx.x;
    int total = Fin * Fout;
    if (idx >= total) return;
    int j    = idx & 15;
    int lane = (idx >> 4) & 31;
    int t    = idx >> 9;            // fragment tile = kt*NT + nt
    int NT   = Fout >> 4;
    int nt   = t % NT;
    int kt   = t / NT;
    int k    = kt * 32 + ((lane & 16) ? 8 : 0) + j + ((j & 8) ? 8 : 0);
    int n    = nt * 16 + (lane & 15);
    P[idx] = (__bf16)W[(size_t)k * Fout + n];
}

// ---------------------------------------------------------------- scatter
// One wave per edge (grid-stride). For edges of relation r:
//   S[dst, :] += h[src, :]   (f32 atomics, L2-resident target)
//   cnt[dst]  += 1           (lane 0)
__global__ __launch_bounds__(256)
void scatter_kernel(const float* __restrict__ h,
                    const int* __restrict__ src, const int* __restrict__ dst,
                    const int* __restrict__ etype, int r, int E, int Fin,
                    float* __restrict__ S, float* __restrict__ cnt) {
    int gid   = blockIdx.x * blockDim.x + threadIdx.x;
    int wid   = gid >> 5;
    int lane  = gid & 31;
    int nwave = (gridDim.x * blockDim.x) >> 5;
    int f4    = Fin >> 2;
    for (int e = wid; e < E; e += nwave) {
        if (etype[e] != r) continue;
        int s = src[e];
        int d = dst[e];
        const float4* xs = (const float4*)(h + (size_t)s * Fin);
        float*        so = S + (size_t)d * Fin;
        for (int c = lane; c < f4; c += 32) {
            float4 v = xs[c];
            float* o = so + (c << 2);
            atomicAdd(o + 0, v.x);
            atomicAdd(o + 1, v.y);
            atomicAdd(o + 2, v.z);
            atomicAdd(o + 3, v.w);
        }
        if (lane == 0) atomicAdd(&cnt[d], 1.0f);
    }
}

// ------------------------------------------------------------------- GEMM
// One wave computes a 16x64 strip of Out[NODES, Fout]: 4 N-tiles sharing one
// A fragment per K-step -> 4 independent v_wmma_f32_16x16x32_bf16 per A load
// (amortizes A traffic + f32->bf16 conversion, hides WMMA->WMMA hazards).
// A rows are f32, scaled by 1/max(cnt,1) (mean agg) then converted to bf16.
// flags: bit0 = init (write bias, don't read Out), bit1 = ReLU epilogue.
__global__ __launch_bounds__(256)
void wmma_gemm_kernel(const float* __restrict__ A, const __bf16* __restrict__ Pw,
                      float* __restrict__ Out, const float* __restrict__ cnt,
                      const float* __restrict__ bias, int Fin, int Fout, int flags) {
    int wid  = (blockIdx.x * blockDim.x + threadIdx.x) >> 5;
    int lane = threadIdx.x & 31;
    int NT   = Fout >> 4;     // N tiles (16 wide)
    int NB   = NT >> 2;       // N blocks (64 wide); all Fout are multiples of 64
    if (wid >= MTILES * NB) return;   // uniform per wave: EXEC stays all-1s
    int nb = wid % NB;
    int mt = wid / NB;
    int nt0 = nb << 2;

    int m = mt * 16 + (lane & 15);    // A-fragment row for this lane
    float scale = 1.0f;
    if (cnt) scale = 1.0f / fmaxf(cnt[m], 1.0f);

    v8f acc0 = {0.f, 0.f, 0.f, 0.f, 0.f, 0.f, 0.f, 0.f};
    v8f acc1 = acc0, acc2 = acc0, acc3 = acc0;

    const int KT = Fin >> 5;
    for (int kt = 0; kt < KT; ++kt) {
        // A layout (16-bit A 16x32): lanes 0-15 hold K = 0..7,16..23;
        // lanes 16-31 hold K = 8..15,24..31 (within the 32-wide K tile).
        int kb = kt * 32 + ((lane & 16) ? 8 : 0);
        const float* row = A + (size_t)m * Fin + kb;
        float4 f0 = *(const float4*)(row);
        float4 f1 = *(const float4*)(row + 4);
        float4 f2 = *(const float4*)(row + 16);
        float4 f3 = *(const float4*)(row + 20);
        v16bf a;
        a[0]  = (__bf16)(f0.x * scale); a[1]  = (__bf16)(f0.y * scale);
        a[2]  = (__bf16)(f0.z * scale); a[3]  = (__bf16)(f0.w * scale);
        a[4]  = (__bf16)(f1.x * scale); a[5]  = (__bf16)(f1.y * scale);
        a[6]  = (__bf16)(f1.z * scale); a[7]  = (__bf16)(f1.w * scale);
        a[8]  = (__bf16)(f2.x * scale); a[9]  = (__bf16)(f2.y * scale);
        a[10] = (__bf16)(f2.z * scale); a[11] = (__bf16)(f2.w * scale);
        a[12] = (__bf16)(f3.x * scale); a[13] = (__bf16)(f3.y * scale);
        a[14] = (__bf16)(f3.z * scale); a[15] = (__bf16)(f3.w * scale);

        // 4 pre-packed B fragments: contiguous 32B per lane, 1KB apart.
        const __bf16* bp = Pw + (((size_t)(kt * NT + nt0)) << 9) + (lane << 4);
        v16bf b0 = *(const v16bf*)(bp);
        v16bf b1 = *(const v16bf*)(bp + 512);
        v16bf b2 = *(const v16bf*)(bp + 1024);
        v16bf b3 = *(const v16bf*)(bp + 1536);

        acc0 = __builtin_amdgcn_wmma_f32_16x16x32_bf16(false, a, false, b0, (short)0, acc0, false, false);
        acc1 = __builtin_amdgcn_wmma_f32_16x16x32_bf16(false, a, false, b1, (short)0, acc1, false, false);
        acc2 = __builtin_amdgcn_wmma_f32_16x16x32_bf16(false, a, false, b2, (short)0, acc2, false, false);
        acc3 = __builtin_amdgcn_wmma_f32_16x16x32_bf16(false, a, false, b3, (short)0, acc3, false, false);
    }

    // C/D layout: VGPR i -> M = i + 8*(lane>=16), N = lane&15.
    int nbase = (nt0 << 4) + (lane & 15);
    int mb    = mt * 16 + ((lane & 16) ? 8 : 0);
    v8f accs[4] = {acc0, acc1, acc2, acc3};
#pragma unroll
    for (int t = 0; t < 4; ++t) {
        int n = nbase + (t << 4);
#pragma unroll
        for (int i = 0; i < 8; ++i) {
            size_t o = (size_t)(mb + i) * Fout + n;
            float v = accs[t][i];
            if (flags & 1) v += bias[n];
            else           v += Out[o];
            if (flags & 2) v = fmaxf(v, 0.0f);
            Out[o] = v;
        }
    }
}

// -------------------------------------------------------------- FC + sigmoid
__global__ __launch_bounds__(256)
void fc_sigmoid_kernel(const float* __restrict__ h, const float* __restrict__ w,
                       const float* __restrict__ b, float* __restrict__ out, int n) {
    int i = blockIdx.x * blockDim.x + threadIdx.x;
    if (i >= n) return;
    float acc = b[0];
    const float4* hr = (const float4*)(h + (size_t)i * 64);
    const float4* wr = (const float4*)w;
#pragma unroll
    for (int j = 0; j < 16; ++j) {
        float4 v  = hr[j];
        float4 ww = wr[j];
        acc += v.x * ww.x + v.y * ww.y + v.z * ww.z + v.w * ww.w;
    }
    out[i] = 1.0f / (1.0f + __expf(-acc));
}

// --------------------------------------------------------------------- host
static inline void zero_buf(float* p, int n, hipStream_t s) {
    int n4 = n >> 2;
    int blocks = (n4 + 255) / 256;
    if (blocks > 1024) blocks = 1024;
    zero4_kernel<<<blocks, 256, 0, s>>>((float4*)p, n4);
}

static void run_layer(hipStream_t stream,
                      const float* in, float* out,
                      const float* wrel, const float* wroot, const float* bias,
                      int Fin, int Fout,
                      const int* src, const int* dst, const int* etype, int E,
                      float* S, float* cnt, __bf16* pk) {
    int NT = Fout >> 4;
    int NB = NT >> 2;
    int packTotal = Fin * Fout;
    int pblocks = (packTotal + 255) / 256;
    pack_w_kernel<<<pblocks, 256, 0, stream>>>(wroot, pk, Fin, Fout);
    for (int r = 0; r < 3; ++r)
        pack_w_kernel<<<pblocks, 256, 0, stream>>>(
            wrel + (size_t)r * packTotal, pk + 65536 * (1 + r), Fin, Fout);

    int strips  = MTILES * NB;
    int gblocks = (strips * 32 + 255) / 256;

    // root transform + bias (init pass)
    wmma_gemm_kernel<<<gblocks, 256, 0, stream>>>(
        in, pk, out, nullptr, bias, Fin, Fout, /*flags=*/1);

    // per-relation: scatter-aggregate then transform-accumulate (ReLU on last)
    for (int r = 0; r < 3; ++r) {
        zero_buf(S,   NODES * Fin, stream);
        zero_buf(cnt, NODES,       stream);
        scatter_kernel<<<2048, 256, 0, stream>>>(in, src, dst, etype, r, E, Fin, S, cnt);
        wmma_gemm_kernel<<<gblocks, 256, 0, stream>>>(
            S, pk + 65536 * (1 + r), out, cnt, nullptr, Fin, Fout,
            (r == 2) ? 2 : 0);
    }
}

extern "C" void kernel_launch(void* const* d_in, const int* in_sizes, int n_in,
                              void* d_out, int out_size, void* d_ws, size_t ws_size,
                              hipStream_t stream) {
    const float* x     = (const float*)d_in[0];
    const int*   eidx  = (const int*)d_in[1];
    const int*   etype = (const int*)d_in[2];
    const int    E     = in_sizes[2];
    const int*   src   = eidx;
    const int*   dst   = eidx + E;

    const float* w1_rel  = (const float*)d_in[3];
    const float* w1_root = (const float*)d_in[4];
    const float* b1      = (const float*)d_in[5];
    const float* w2_rel  = (const float*)d_in[6];
    const float* w2_root = (const float*)d_in[7];
    const float* b2      = (const float*)d_in[8];
    const float* w3_rel  = (const float*)d_in[9];
    const float* w3_root = (const float*)d_in[10];
    const float* b3      = (const float*)d_in[11];
    const float* w4_rel  = (const float*)d_in[12];
    const float* w4_root = (const float*)d_in[13];
    const float* b4      = (const float*)d_in[14];
    const float* fc_w    = (const float*)d_in[15];
    const float* fc_b    = (const float*)d_in[16];

    // Workspace carve (all 64B aligned: NODES*256*4 and NODES*4 are multiples of 64)
    float* hA  = (float*)d_ws;             // 50000 x 256
    float* hB  = hA + (size_t)NODES * 256; // 50000 x 256
    float* S   = hB + (size_t)NODES * 256; // 50000 x 256 scatter accumulator
    float* cnt = S  + (size_t)NODES * 256; // 50000
    __bf16* pk = (__bf16*)(cnt + NODES);   // 4 slabs of 256*256 bf16 packed weights

    run_layer(stream, x,  hA, w1_rel, w1_root, b1, 128, 256, src, dst, etype, E, S, cnt, pk);
    run_layer(stream, hA, hB, w2_rel, w2_root, b2, 256, 128, src, dst, etype, E, S, cnt, pk);
    run_layer(stream, hB, hA, w3_rel, w3_root, b3, 128,  64, src, dst, etype, E, S, cnt, pk);
    run_layer(stream, hA, hB, w4_rel, w4_root, b4,  64,  64, src, dst, etype, E, S, cnt, pk);

    fc_sigmoid_kernel<<<(NODES + 255) / 256, 256, 0, stream>>>(
        hB, fc_w, fc_b, (float*)d_out, NODES);
}
